// LSTM_75883482186342
// MI455X (gfx1250) — matrix operand order repaired and checked by reading, not compiled, not run on hardware
//
#include <hip/hip_runtime.h>

typedef __attribute__((ext_vector_type(16))) __bf16 v16bf;
typedef __attribute__((ext_vector_type(8)))  float  v8f;

constexpr int NWG   = 16;    // workgroups; each owns 32 columns of h
constexpr int TPB   = 256;   // 8 wave32 per WG
constexpr int BDIM  = 128;
constexpr int HDIM  = 512;
constexpr int SEQ   = 512;
constexpr int STEPS = SEQ - 1;   // 511
constexpr int CDIM  = 10;
constexpr int KC    = HDIM / 32; // 16 K-chunks of 32
constexpr int COLS  = HDIM / NWG; // 32 cols per WG
constexpr int WLDS_ELEMS = 4 * 2 * KC * 32 * 16; // 65536 bf16 = 128KB

union Frag { v16bf v; uint4 u[2]; };

__device__ __forceinline__ float sigmoidf_(float x) { return 1.0f / (1.0f + __expf(-x)); }

// Reset barrier counters and zero the step-0 hidden-state buffer (h=0, c=0 init).
__global__ void lstm_init(int* bar, __bf16* hbuf) {
  int i = blockIdx.x * blockDim.x + threadIdx.x;
  if (i < 2) bar[i] = 0;
  for (int k = i; k < 2 * BDIM * HDIM; k += gridDim.x * blockDim.x)
    hbuf[k] = (__bf16)0.0f;
}

__global__ void __launch_bounds__(TPB, 1)
lstm_kernel(const float* __restrict__ x,
            const float* __restrict__ Wgx, const float* __restrict__ Wix,
            const float* __restrict__ Wfx, const float* __restrict__ Wox,
            const float* __restrict__ Wgh, const float* __restrict__ Wih,
            const float* __restrict__ Wfh, const float* __restrict__ Woh,
            const float* __restrict__ bg,  const float* __restrict__ bi,
            const float* __restrict__ bf_, const float* __restrict__ bo,
            const float* __restrict__ outW, const float* __restrict__ outB,
            __bf16* __restrict__ hbuf,   // [2][BDIM][HDIM] bf16, double-buffered
            float*  __restrict__ hfin,   // [BDIM][HDIM] f32 (final step only)
            int*    bar,                 // [0]=count, [1]=generation
            float*  __restrict__ out)    // [BDIM][CDIM]
{
  extern __shared__ char smem[];
  __bf16* w_lds = (__bf16*)smem;  // swizzled B-fragments: [g][ct][kc][lane][16]

  const int tid  = threadIdx.x;
  const int wg   = blockIdx.x;
  const int wave = tid >> 5;
  const int lane = tid & 31;
  const int lr   = lane & 15;
  const int hi   = lane >> 4;

  const float* Wh[4] = {Wgh, Wih, Wfh, Woh};
  const float* Wx[4] = {Wgx, Wix, Wfx, Wox};
  const float* Bb[4] = {bg, bi, bf_, bo};

  // ---- One-time: stage this WG's weight slice into LDS, bf16, WMMA-B-swizzled.
  // B tile layout (16x16x32 bf16): lanes 0-15 -> N=lane, K={0..7,16..23}; lanes 16-31 -> N=lane-16, K={8..15,24..31}.
  for (int idx = tid; idx < WLDS_ELEMS; idx += TPB) {
    int j  = idx & 15;          // K within lane's 16-value run
    int l  = (idx >> 4) & 31;   // lane
    int kc = (idx >> 9) & 15;   // K chunk of 32
    int ct = (idx >> 13) & 1;   // col tile (16 cols)
    int g  = (idx >> 14) & 3;   // gate
    int col = wg * COLS + ct * 16 + (l & 15);
    int K   = kc * 32 + ((l >> 4) << 3) + (j & 7) + ((j >> 3) << 4);
    w_lds[idx] = (__bf16)Wh[g][K * HDIM + col];
  }
  __syncthreads();

  // Per-lane constants: rank-1 input weights + biases for this lane's column.
  float wx_c[4][2], b_c[4][2];
#pragma unroll
  for (int g = 0; g < 4; ++g)
#pragma unroll
    for (int ct = 0; ct < 2; ++ct) {
      int col = wg * COLS + ct * 16 + lr;
      wx_c[g][ct] = Wx[g][col];
      b_c[g][ct]  = Bb[g][col];
    }

  const int rt = wave;  // this wave's 16-row tile of the batch
  v8f cstate[2];
#pragma unroll
  for (int ct = 0; ct < 2; ++ct)
#pragma unroll
    for (int r = 0; r < 8; ++r) cstate[ct][r] = 0.0f;

  for (int step = 0; step < STEPS; ++step) {
    const __bf16* hsrc = hbuf + (size_t)(step & 1) * (BDIM * HDIM);
    __bf16*       hdst = hbuf + (size_t)((step + 1) & 1) * (BDIM * HDIM);

    v8f acc[4][2];
#pragma unroll
    for (int g = 0; g < 4; ++g)
#pragma unroll
      for (int ct = 0; ct < 2; ++ct)
#pragma unroll
        for (int r = 0; r < 8; ++r) acc[g][ct][r] = 0.0f;

    // A tile (16x32 bf16): lane = M; lanes 0-15 hold K {0..7,16..23}, 16-31 hold {8..15,24..31}.
    const int arow = rt * 16 + lr;
    for (int kc = 0; kc < KC; ++kc) {
      Frag a;
      const __bf16* ap = hsrc + arow * HDIM + kc * 32 + hi * 8;
      a.u[0] = *(const uint4*)(ap);
      a.u[1] = *(const uint4*)(ap + 16);
#pragma unroll
      for (int g = 0; g < 4; ++g) {
#pragma unroll
        for (int ct = 0; ct < 2; ++ct) {
          Frag b;
          const __bf16* bp = w_lds + ((((g * 2 + ct) * KC) + kc) << 9) + lane * 16;
          b.u[0] = *(const uint4*)(bp);
          b.u[1] = *(const uint4*)(bp + 8);
          acc[g][ct] = __builtin_amdgcn_wmma_f32_16x16x32_bf16(
              false, a.v, false, b.v, (short)0, acc[g][ct], false, false);
        }
      }
    }

    // Elementwise gate math, f32, wave-local (this wave owns all 4 gates for its rows).
    // D layout: vgpr r -> M = hi*8 + r, N = lr.
#pragma unroll
    for (int ct = 0; ct < 2; ++ct) {
      const int col = wg * COLS + ct * 16 + lr;
#pragma unroll
      for (int r = 0; r < 8; ++r) {
        const int brow = rt * 16 + hi * 8 + r;
        const float xv = x[brow * SEQ + step];
        float gp = acc[0][ct][r] + xv * wx_c[0][ct] + b_c[0][ct];
        float ip = acc[1][ct][r] + xv * wx_c[1][ct] + b_c[1][ct];
        float fp = acc[2][ct][r] + xv * wx_c[2][ct] + b_c[2][ct];
        float op = acc[3][ct][r] + xv * wx_c[3][ct] + b_c[3][ct];
        float gg = tanhf(gp);
        float ii = sigmoidf_(ip);
        float ff = sigmoidf_(fp);
        float oo = sigmoidf_(op);
        float cn = gg * ii + cstate[ct][r] * ff;
        cstate[ct][r] = cn;
        float hh = tanhf(cn) * oo;
        hdst[brow * HDIM + col] = (__bf16)hh;
        if (step == STEPS - 1) hfin[brow * HDIM + col] = hh;
      }
    }

    // Sense-free grid barrier: monotone generation counter (reset by init kernel).
    __syncthreads();
    if (tid == 0) {
      __threadfence();
      if (atomicAdd(&bar[0], 1) == NWG - 1) {
        bar[0] = 0;
        __threadfence();
        atomicAdd(&bar[1], 1);
      } else {
        while (__hip_atomic_load(&bar[1], __ATOMIC_ACQUIRE, __HIP_MEMORY_SCOPE_AGENT) <= step)
          __builtin_amdgcn_s_sleep(1);
      }
    }
    __syncthreads();
  }

  // Final projection h @ outW + outB by WG 0 (1.3 MFLOP; trivial).
  if (wg == 0) {
    for (int idx = tid; idx < BDIM * CDIM; idx += TPB) {
      int b = idx / CDIM, c = idx % CDIM;
      float s = outB[c];
      for (int k = 0; k < HDIM; ++k) s += hfin[b * HDIM + k] * outW[k * CDIM + c];
      out[idx] = s;
    }
  }
}

extern "C" void kernel_launch(void* const* d_in, const int* in_sizes, int n_in,
                              void* d_out, int out_size, void* d_ws, size_t ws_size,
                              hipStream_t stream) {
  (void)in_sizes; (void)n_in; (void)out_size; (void)ws_size;
  const float* x   = (const float*)d_in[0];
  const float* Wgx = (const float*)d_in[1];
  const float* Wix = (const float*)d_in[2];
  const float* Wfx = (const float*)d_in[3];
  const float* Wox = (const float*)d_in[4];
  const float* Wgh = (const float*)d_in[5];
  const float* Wih = (const float*)d_in[6];
  const float* Wfh = (const float*)d_in[7];
  const float* Woh = (const float*)d_in[8];
  const float* bg  = (const float*)d_in[9];
  const float* bi  = (const float*)d_in[10];
  const float* bf_ = (const float*)d_in[11];
  const float* bo  = (const float*)d_in[12];
  const float* oW  = (const float*)d_in[13];
  const float* oB  = (const float*)d_in[14];

  char* ws = (char*)d_ws;
  int*    bar  = (int*)ws;                                   // 256 B reserved
  __bf16* hbuf = (__bf16*)(ws + 256);                        // 2*128*512*2 = 256 KB
  float*  hfin = (float*)(ws + 256 + 2 * BDIM * HDIM * 2);   // 256 KB
  float*  out  = (float*)d_out;

  lstm_init<<<8, 256, 0, stream>>>(bar, hbuf);

  const size_t shmem = WLDS_ELEMS * sizeof(__bf16);          // 128 KB LDS
  (void)hipFuncSetAttribute((const void*)lstm_kernel,
                            hipFuncAttributeMaxDynamicSharedMemorySize, (int)shmem);
  lstm_kernel<<<NWG, TPB, shmem, stream>>>(x, Wgx, Wix, Wfx, Wox,
                                           Wgh, Wih, Wfh, Woh,
                                           bg, bi, bf_, bo, oW, oB,
                                           hbuf, hfin, bar, out);
}
